// AttentiveModel_2439541424837
// MI455X (gfx1250) — matrix operand order repaired and checked by reading, not compiled, not run on hardware
//
#include <hip/hip_runtime.h>
#include <math.h>

#define B_   32
#define TI_  128
#define TO_  64
#define VO_  32000
#define E_   256
#define H_   512
#define A_   512

typedef __attribute__((ext_vector_type(16))) __bf16 v16bf;
typedef __attribute__((ext_vector_type(8)))  __bf16 v8bf;
typedef __attribute__((ext_vector_type(8)))  float  v8f;
typedef __attribute__((ext_vector_type(4)))  int    i32x4;

union FragA { v16bf v; v8bf h[2]; };

#if defined(__has_builtin)
#if __has_builtin(__builtin_amdgcn_global_load_async_to_lds_b128)
#define HAVE_ASYNC_LDS 1
#endif
#endif

__device__ __forceinline__ unsigned short f2bf(float f) {
  unsigned int u = __float_as_uint(f);
  u += 0x7FFFu + ((u >> 16) & 1u);           // round-to-nearest-even
  return (unsigned short)(u >> 16);
}
__device__ __forceinline__ float bf2f(unsigned short h) {
  return __uint_as_float(((unsigned int)h) << 16);
}
__device__ __forceinline__ float sigm(float x) { return 1.0f / (1.0f + expf(-x)); }

// ---------------------------------------------------------------------------
// bf16 WMMA GEMM:  C[M,N](fp32) = A[M,K](bf16) * W[N,K]^T(bf16) + bias
// Block = 4 waves = one 16-row x 256-col tile. The 16xK A tile (contiguous in
// memory) is staged into LDS once per block — via async-to-LDS when available
// — and A fragments are then ds_loads shared by all 4 waves. B fragments are
// ping-pong double-buffered from global so WMMAs overlap outstanding loads.
// Requires: M%16==0, N%256==0, K%64==0 (true for every GEMM in this model).
// ---------------------------------------------------------------------------
__global__ void __launch_bounds__(128)
k_gemm_bf16(const unsigned short* __restrict__ A,
            const unsigned short* __restrict__ W,
            const float* __restrict__ bias,
            float* __restrict__ C,
            int M, int N, int K, long long ldc)
{
  __shared__ unsigned short sA[16 * 1536];            // up to 48 KB of 320 KB LDS
  const int tid = threadIdx.x;
  const int nb4 = N >> 8;
  const int mt  = blockIdx.x / nb4;
  const int nb  = blockIdx.x - mt * nb4;

  // ---- stage 16xK A tile (rows mt*16..mt*16+15 are contiguous: 32*K bytes) ----
  {
    const unsigned short* gsrc = A + (size_t)mt * 16 * K;
    const int elems = 16 * K;
#if HAVE_ASYNC_LDS
    __attribute__((address_space(1))) i32x4* g =
        (__attribute__((address_space(1))) i32x4*)(void*)gsrc;
    __attribute__((address_space(3))) i32x4* l =
        (__attribute__((address_space(3))) i32x4*)(void*)sA;
    const int nvec = (elems * 2) / 16;                // 16-byte packets
    for (int v = tid; v < nvec; v += 128)
      __builtin_amdgcn_global_load_async_to_lds_b128(g + v, l + v, 0, 0);
#if __has_builtin(__builtin_amdgcn_s_wait_asynccnt)
    __builtin_amdgcn_s_wait_asynccnt(0);
#else
    asm volatile("s_wait_asynccnt 0" ::: "memory");
#endif
#else
    for (int e = tid * 8; e < elems; e += 128 * 8)
      *(v8bf*)&sA[e] = *(const v8bf*)(gsrc + e);
#endif
    __syncthreads();
  }

  const int lane = tid & 31;
  const int wv   = tid >> 5;                          // 0..3
  const int l15  = lane & 15;
  const int hi   = lane >> 4;
  const int kba  = hi * 16;                           // A-frag K base in 32-chunk
  const int kbb  = hi * 16;                           // B-frag K base in 32-chunk
  const size_t colbase = (size_t)nb * 256 + (size_t)wv * 64;

  const unsigned short* aL  = sA + l15 * K;           // LDS row for this lane
  const unsigned short* wp0 = W + (colbase + l15) * (size_t)K;

  v8f acc[4] = {};
  FragA fa0, fa1;
  v16bf fb0[4], fb1[4];

  auto loadA = [&](int kk, FragA& fa) {
    fa.h[0] = *(const v8bf*)(aL + kk + kba);          // ds_load_b128
    fa.h[1] = *(const v8bf*)(aL + kk + kba + 8);
  };
  auto loadB = [&](int kk, v16bf* fb) {
#pragma unroll
    for (int j = 0; j < 4; ++j)
      fb[j] = *(const v16bf*)(wp0 + (size_t)j * 16 * K + kk + kbb);
  };
  auto mma4 = [&](FragA& fa, v16bf* fb) {
#pragma unroll
    for (int j = 0; j < 4; ++j)
      acc[j] = __builtin_amdgcn_wmma_f32_16x16x32_bf16(
          false, fa.v, false, fb[j], (short)0, acc[j], false, false);
  };

  loadA(0, fa0); loadB(0, fb0);
  for (int kk = 0; kk < K; kk += 64) {                // K%64==0: clean ping-pong
    loadA(kk + 32, fa1); loadB(kk + 32, fb1);
    if (kk + 256 < K) __builtin_prefetch(wp0 + kk + 256, 0, 1);
    mma4(fa0, fb0);
    if (kk + 64 < K) { loadA(kk + 64, fa0); loadB(kk + 64, fb0); }
    mma4(fa1, fb1);
  }

  const int r0 = mt * 16 + hi * 8;
#pragma unroll
  for (int j = 0; j < 4; ++j) {
    const size_t col = colbase + (size_t)j * 16 + l15;
    const float bv = bias ? bias[col] : 0.0f;
#pragma unroll
    for (int r = 0; r < 8; ++r)
      C[(size_t)(r0 + r) * (size_t)ldc + col] = acc[j][r] + bv;
  }
}

// ---------------------------------------------------------------------------
// GRU scan, one direction. Single workgroup of 32 waves; hidden state [B,H]
// lives in LDS as bf16. Wave w owns output columns [16w,16w+16) for all three
// gates so the nonlinearity is wave-local. K-loop is ping-pong double-buffered
// (Whh from global / L2, A-frags from LDS).
// ---------------------------------------------------------------------------
__global__ void __launch_bounds__(1024)
k_gru_scan(const float* __restrict__ xp,              // [B,TI,3H] (bih included)
           const unsigned short* __restrict__ Whh,    // [3H,H] bf16
           const float* __restrict__ bhh,             // [3H]
           float* __restrict__ hs,                    // [B,TI,H] output
           int reverse)
{
  __shared__ unsigned short sh[B_ * H_];              // 32 KB
  const int tid = threadIdx.x;
  for (int i = tid; i < B_ * H_; i += 1024) sh[i] = 0;
  __syncthreads();

  const int wave = tid >> 5;
  const int lane = tid & 31;
  const int l15  = lane & 15;
  const int hi   = lane >> 4;
  const int colL = wave * 16 + l15;
  const int kba  = hi * 16;
  const int kbb  = hi * 16;

  const float bR = bhh[colL], bZ = bhh[H_ + colL], bN = bhh[2 * H_ + colL];
  const unsigned short* wR0 = Whh + (size_t)(colL) * H_;
  const unsigned short* wZ0 = Whh + (size_t)(H_ + colL) * H_;
  const unsigned short* wN0 = Whh + (size_t)(2 * H_ + colL) * H_;

  for (int t = 0; t < TI_; ++t) {
    const int tt = reverse ? (TI_ - 1 - t) : t;
    float hnew[2][8];
#pragma unroll
    for (int mt = 0; mt < 2; ++mt) {
      v8f aR = {}, aZ = {}, aN = {};
      const int arow = mt * 16 + l15;

      auto loadAsh = [&](int kk, FragA& fa) {
        fa.h[0] = *(const v8bf*)(&sh[arow * H_ + kk + kba]);
        fa.h[1] = *(const v8bf*)(&sh[arow * H_ + kk + kba + 8]);
      };
      auto loadW = [&](int kk, v16bf& fR, v16bf& fZ, v16bf& fN) {
        fR = *(const v16bf*)(wR0 + kk + kbb);
        fZ = *(const v16bf*)(wZ0 + kk + kbb);
        fN = *(const v16bf*)(wN0 + kk + kbb);
      };
      auto mma3 = [&](FragA& fa, v16bf& fR, v16bf& fZ, v16bf& fN) {
        aR = __builtin_amdgcn_wmma_f32_16x16x32_bf16(false, fa.v, false, fR, (short)0, aR, false, false);
        aZ = __builtin_amdgcn_wmma_f32_16x16x32_bf16(false, fa.v, false, fZ, (short)0, aZ, false, false);
        aN = __builtin_amdgcn_wmma_f32_16x16x32_bf16(false, fa.v, false, fN, (short)0, aN, false, false);
      };

      FragA fa0, fa1;
      v16bf R0, Z0, N0, R1, Z1, N1;
      loadAsh(0, fa0); loadW(0, R0, Z0, N0);
      for (int kk = 0; kk < H_; kk += 64) {           // H%64==0
        loadAsh(kk + 32, fa1); loadW(kk + 32, R1, Z1, N1);
        mma3(fa0, R0, Z0, N0);
        if (kk + 64 < H_) { loadAsh(kk + 64, fa0); loadW(kk + 64, R0, Z0, N0); }
        mma3(fa1, R1, Z1, N1);
      }

      const int r0 = mt * 16 + hi * 8;
#pragma unroll
      for (int r = 0; r < 8; ++r) {
        const int b = r0 + r;
        const size_t xb = ((size_t)b * TI_ + tt) * (3 * H_) + colL;
        float rg = sigm(xp[xb] + aR[r] + bR);
        float zg = sigm(xp[xb + H_] + aZ[r] + bZ);
        float ng = tanhf(xp[xb + 2 * H_] + rg * (aN[r] + bN));
        float hp = bf2f(sh[b * H_ + colL]);
        hnew[mt][r] = (1.0f - zg) * ng + zg * hp;
      }
    }
    __syncthreads();   // all waves done READING h
#pragma unroll
    for (int mt = 0; mt < 2; ++mt) {
      const int r0 = mt * 16 + hi * 8;
#pragma unroll
      for (int r = 0; r < 8; ++r) {
        const int b = r0 + r;
        sh[b * H_ + colL] = f2bf(hnew[mt][r]);
        hs[((size_t)b * TI_ + tt) * H_ + colL] = hnew[mt][r];
      }
    }
    __syncthreads();   // h_new visible before next step
  }
}

// ---------------------------------------------------------------------------
// Elementwise / small kernels
// ---------------------------------------------------------------------------
__global__ void k_f32_to_bf16(const float* __restrict__ s, unsigned short* __restrict__ d, int n) {
  for (int i = blockIdx.x * blockDim.x + threadIdx.x; i < n; i += gridDim.x * blockDim.x)
    d[i] = f2bf(s[i]);
}

__global__ void k_gather_x(const int* __restrict__ inp, const float* __restrict__ emb,
                           unsigned short* __restrict__ x) {
  int i = blockIdx.x * blockDim.x + threadIdx.x;
  if (i >= B_ * TI_ * E_) return;
  int bt = i >> 8, d = i & (E_ - 1);
  x[i] = f2bf(emb[(size_t)inp[bt] * E_ + d]);
}

__global__ void k_concat_enc(const float* __restrict__ ef, const float* __restrict__ eb,
                             float* __restrict__ enc_f, unsigned short* __restrict__ enc_bf) {
  int i = blockIdx.x * blockDim.x + threadIdx.x;
  if (i >= B_ * TI_ * 2 * H_) return;
  int bt = i >> 10, d = i & 1023;
  float v = (d < H_) ? ef[(size_t)bt * H_ + d] : eb[(size_t)bt * H_ + d - H_];
  enc_f[i] = v;
  enc_bf[i] = f2bf(v);
}

__global__ void k_last(const int* __restrict__ inp, const unsigned short* __restrict__ enc_bf,
                       unsigned short* __restrict__ last_bf) {
  int b = blockIdx.x;
  __shared__ int s_len;
  if (threadIdx.x == 0) {
    int c = 0;
    for (int t = 0; t < TI_; ++t) c += (inp[b * TI_ + t] != 0);
    s_len = c < (TI_ - 1) ? c : (TI_ - 1);
  }
  __syncthreads();
  for (int d = threadIdx.x; d < 2 * H_; d += blockDim.x)
    last_bf[b * 2 * H_ + d] = enc_bf[((size_t)b * TI_ + s_len) * (2 * H_) + d];
}

// Bahdanau attention for one decoder state: scores -> masked softmax -> context
__global__ void __launch_bounds__(128)
k_attend(const float* __restrict__ enc_proj,   // [B,TI,A]
         const float* __restrict__ dp,         // [B,A]
         const float* __restrict__ attn_v,     // [A]
         const float* __restrict__ vb,         // scalar
         const int* __restrict__ inp,          // [B,TI] for mask
         const float* __restrict__ enc_f,      // [B,TI,2H]
         float* __restrict__ a_out)            // [B,2H]
{
  int b = blockIdx.x, tid = threadIdx.x;
  __shared__ float s_dp[A_], s_v[A_], s_sc[TI_];
  for (int i = tid; i < A_; i += 128) { s_dp[i] = dp[b * A_ + i]; s_v[i] = attn_v[i]; }
  __syncthreads();
  {
    int t = tid;
    float s = vb[0];
    const float* ep = enc_proj + ((size_t)b * TI_ + t) * A_;
    for (int a = 0; a < A_; ++a) s += tanhf(ep[a] + s_dp[a]) * s_v[a];
    if (inp[b * TI_ + t] == 0) s = -INFINITY;
    s_sc[t] = s;
  }
  __syncthreads();
  float m = -INFINITY;
  for (int i = 0; i < TI_; ++i) m = fmaxf(m, s_sc[i]);
  float sum = 0.0f;
  for (int i = 0; i < TI_; ++i) sum += expf(s_sc[i] - m);
  __syncthreads();
  float pr = expf(s_sc[tid] - m) / sum;
  __syncthreads();
  s_sc[tid] = pr;
  __syncthreads();
  for (int d = tid; d < 2 * H_; d += 128) {
    float acc = 0.0f;
    for (int t = 0; t < TI_; ++t)
      acc += s_sc[t] * enc_f[((size_t)b * TI_ + t) * (2 * H_) + d];
    a_out[b * 2 * H_ + d] = acc;
  }
}

__global__ void k_xcat(const int* __restrict__ out_tok, int s,
                       const float* __restrict__ emb_out, const float* __restrict__ a_f,
                       unsigned short* __restrict__ xcat) {
  int i = blockIdx.x * blockDim.x + threadIdx.x;
  const int W = E_ + 2 * H_;                     // 1280
  if (i >= B_ * W) return;
  int b = i / W, d = i - b * W;
  float v = (d < E_) ? emb_out[(size_t)out_tok[b * TO_ + s] * E_ + d]
                     : a_f[b * 2 * H_ + d - E_];
  xcat[i] = f2bf(v);
}

__global__ void k_dec_gates(const float* __restrict__ xg, const float* __restrict__ hg,
                            float* __restrict__ h_f, unsigned short* __restrict__ h_bf) {
  int i = blockIdx.x * blockDim.x + threadIdx.x;
  if (i >= B_ * H_) return;
  int b = i >> 9, c = i & (H_ - 1);
  size_t g = (size_t)b * 3 * H_ + c;
  float r = sigm(xg[g] + hg[g]);
  float z = sigm(xg[g + H_] + hg[g + H_]);
  float n = tanhf(xg[g + 2 * H_] + r * hg[g + 2 * H_]);
  float hn = (1.0f - z) * n + z * h_f[i];
  h_f[i] = hn;
  h_bf[i] = f2bf(hn);
}

__global__ void k_hc(const float* __restrict__ h_f, const float* __restrict__ a_f,
                     unsigned short* __restrict__ hc) {
  int i = blockIdx.x * blockDim.x + threadIdx.x;
  if (i >= B_ * 3 * H_) return;
  int b = i / (3 * H_), d = i - b * 3 * H_;
  float v = (d < H_) ? h_f[b * H_ + d] : a_f[b * 2 * H_ + d - H_];
  hc[i] = f2bf(v);
}

__global__ void k_first_logits(float* __restrict__ out) {
  int i = blockIdx.x * blockDim.x + threadIdx.x;
  if (i >= B_ * VO_) return;
  int b = i / VO_, v = i - b * VO_;
  out[(size_t)b * TO_ * VO_ + v] = logf((v == 1 ? 1.0f : 0.0f) + 1e-9f);
}

// ---------------------------------------------------------------------------
// Host side
// ---------------------------------------------------------------------------
static inline void gemm(const unsigned short* A, const unsigned short* W, const float* bias,
                        float* C, int M, int N, int K, long long ldc, hipStream_t st) {
  int blocks = (M / 16) * (N / 256);
  k_gemm_bf16<<<blocks, 128, 0, st>>>(A, W, bias, C, M, N, K, ldc);
}
static inline void cvt(const float* s, unsigned short* d, int n, hipStream_t st) {
  int b = (n + 255) / 256; if (b > 4096) b = 4096;
  k_f32_to_bf16<<<b, 256, 0, st>>>(s, d, n);
}

extern "C" void kernel_launch(void* const* d_in, const int* in_sizes, int n_in,
                              void* d_out, int out_size, void* d_ws, size_t ws_size,
                              hipStream_t stream)
{
  (void)in_sizes; (void)n_in; (void)out_size; (void)ws_size;
  const int*   inp        = (const int*)  d_in[0];
  const int*   out_tok    = (const int*)  d_in[1];
  const float* emb_inp    = (const float*)d_in[2];
  const float* emb_out    = (const float*)d_in[3];
  const float* enc_Wih_f  = (const float*)d_in[4];
  const float* enc_Whh_f  = (const float*)d_in[5];
  const float* enc_bih_f  = (const float*)d_in[6];
  const float* enc_bhh_f  = (const float*)d_in[7];
  const float* enc_Wih_b  = (const float*)d_in[8];
  const float* enc_Whh_b  = (const float*)d_in[9];
  const float* enc_bih_b  = (const float*)d_in[10];
  const float* enc_bhh_b  = (const float*)d_in[11];
  const float* dec_startW = (const float*)d_in[12];
  const float* dec_startb = (const float*)d_in[13];
  const float* dec_Wih    = (const float*)d_in[14];
  const float* dec_Whh    = (const float*)d_in[15];
  const float* dec_bih    = (const float*)d_in[16];
  const float* dec_bhh    = (const float*)d_in[17];
  const float* attn_Wenc  = (const float*)d_in[18];
  const float* attn_benc  = (const float*)d_in[19];
  const float* attn_Wdec  = (const float*)d_in[20];
  const float* attn_bdec  = (const float*)d_in[21];
  const float* attn_v     = (const float*)d_in[22];
  const float* attn_vb    = (const float*)d_in[23];
  const float* logits_W   = (const float*)d_in[24];
  const float* logits_b   = (const float*)d_in[25];
  float* out = (float*)d_out;

  // ---- workspace carve ----
  char* p = (char*)d_ws;
  auto alloc = [&](size_t bytes) -> void* {
    void* r = (void*)p; p += (bytes + 255) & ~(size_t)255; return r;
  };
  auto ua = [&](size_t n) { return (unsigned short*)alloc(n * 2); };
  auto fa = [&](size_t n) { return (float*)alloc(n * 4); };

  unsigned short* wih_f  = ua(1536 * 256);
  unsigned short* wih_b  = ua(1536 * 256);
  unsigned short* whh_f  = ua(1536 * 512);
  unsigned short* whh_b  = ua(1536 * 512);
  unsigned short* wstart = ua(512 * 1024);
  unsigned short* wdecih = ua(1536 * 1280);
  unsigned short* wdechh = ua(1536 * 512);
  unsigned short* wenc   = ua(512 * 1024);
  unsigned short* wdec   = ua(512 * 512);
  unsigned short* wlog   = ua((size_t)VO_ * 1536);      // 98.3 MB -> L2-resident
  unsigned short* x_bf   = ua((size_t)B_ * TI_ * E_);
  unsigned short* enc_bf = ua((size_t)B_ * TI_ * 2 * H_);
  unsigned short* last_bf= ua(B_ * 2 * H_);
  unsigned short* h_bf   = ua(B_ * H_);
  unsigned short* xcat_bf= ua(B_ * (E_ + 2 * H_));
  unsigned short* hc_bf  = ua(B_ * 3 * H_);
  float* xp_f   = fa((size_t)B_ * TI_ * 3 * H_);
  float* xp_b   = fa((size_t)B_ * TI_ * 3 * H_);
  float* ef     = fa((size_t)B_ * TI_ * H_);
  float* eb     = fa((size_t)B_ * TI_ * H_);
  float* enc_f  = fa((size_t)B_ * TI_ * 2 * H_);
  float* eproj  = fa((size_t)B_ * TI_ * A_);
  float* h_f    = fa(B_ * H_);
  float* a_f    = fa(B_ * 2 * H_);
  float* dp_f   = fa(B_ * A_);
  float* xg_f   = fa(B_ * 3 * H_);
  float* hg_f   = fa(B_ * 3 * H_);

  // ---- weight conversion fp32 -> bf16 ----
  cvt(enc_Wih_f, wih_f, 1536 * 256, stream);
  cvt(enc_Wih_b, wih_b, 1536 * 256, stream);
  cvt(enc_Whh_f, whh_f, 1536 * 512, stream);
  cvt(enc_Whh_b, whh_b, 1536 * 512, stream);
  cvt(dec_startW, wstart, 512 * 1024, stream);
  cvt(dec_Wih, wdecih, 1536 * 1280, stream);
  cvt(dec_Whh, wdechh, 1536 * 512, stream);
  cvt(attn_Wenc, wenc, 512 * 1024, stream);
  cvt(attn_Wdec, wdec, 512 * 512, stream);
  cvt(logits_W, wlog, VO_ * 1536, stream);

  // ---- encoder ----
  k_gather_x<<<(B_ * TI_ * E_ + 255) / 256, 256, 0, stream>>>(inp, emb_inp, x_bf);
  gemm(x_bf, wih_f, enc_bih_f, xp_f, B_ * TI_, 3 * H_, E_, 3 * H_, stream);
  gemm(x_bf, wih_b, enc_bih_b, xp_b, B_ * TI_, 3 * H_, E_, 3 * H_, stream);
  k_gru_scan<<<1, 1024, 0, stream>>>(xp_f, whh_f, enc_bhh_f, ef, 0);
  k_gru_scan<<<1, 1024, 0, stream>>>(xp_b, whh_b, enc_bhh_b, eb, 1);
  k_concat_enc<<<(B_ * TI_ * 2 * H_ + 255) / 256, 256, 0, stream>>>(ef, eb, enc_f, enc_bf);
  k_last<<<B_, 256, 0, stream>>>(inp, enc_bf, last_bf);

  // dec0 = last @ dec_start_W^T + b
  gemm(last_bf, wstart, dec_startb, h_f, B_, H_, 2 * H_, H_, stream);
  cvt(h_f, h_bf, B_ * H_, stream);
  // enc_proj
  gemm(enc_bf, wenc, attn_benc, eproj, B_ * TI_, A_, 2 * H_, A_, stream);
  // a0 = attend(dec0)
  gemm(h_bf, wdec, attn_bdec, dp_f, B_, A_, H_, A_, stream);
  k_attend<<<B_, 128, 0, stream>>>(eproj, dp_f, attn_v, attn_vb, inp, enc_f, a_f);

  // first-token logits = log(onehot(BOS) + 1e-9)
  k_first_logits<<<(B_ * VO_ + 255) / 256, 256, 0, stream>>>(out);

  // ---- decoder: 63 sequential steps ----
  for (int s = 0; s < TO_ - 1; ++s) {
    k_xcat<<<(B_ * (E_ + 2 * H_) + 255) / 256, 256, 0, stream>>>(out_tok, s, emb_out, a_f, xcat_bf);
    gemm(xcat_bf, wdecih, dec_bih, xg_f, B_, 3 * H_, E_ + 2 * H_, 3 * H_, stream);
    gemm(h_bf, wdechh, dec_bhh, hg_f, B_, 3 * H_, H_, 3 * H_, stream);
    k_dec_gates<<<(B_ * H_ + 255) / 256, 256, 0, stream>>>(xg_f, hg_f, h_f, h_bf);
    gemm(h_bf, wdec, attn_bdec, dp_f, B_, A_, H_, A_, stream);
    k_attend<<<B_, 128, 0, stream>>>(eproj, dp_f, attn_v, attn_vb, inp, enc_f, a_f);
    k_hc<<<(B_ * 3 * H_ + 255) / 256, 256, 0, stream>>>(h_f, a_f, hc_bf);
    // logits for token s+1 straight into d_out ([B,TO,VO], row stride TO*VO)
    gemm(hc_bf, wlog, logits_b, out + (size_t)(s + 1) * VO_,
         B_, VO_, 3 * H_, (long long)TO_ * VO_, stream);
  }
}
